// KnnAttentionPool_35347580846877
// MI455X (gfx1250) — matrix operand order repaired and checked by reading, not compiled, not run on hardware
//
#include <hip/hip_runtime.h>
#include <hip/hip_bf16.h>

typedef __attribute__((ext_vector_type(16))) _Float16 v16h;
typedef __attribute__((ext_vector_type(8)))  float    v8f;

#define M_Q   16384
#define N_CTX 131072
#define KNN   16
#define CDIM  128
#define GRP   8
#define WPITCH (CDIM + 8)          // 136 halves per row (272B) breaks bank stride
#define WIMG_HALVES (CDIM * WPITCH) // 17408 halves = 34816 B per 128x128 image

// 1/sqrt(1 + 1e-5)
#define BN_INV 0.9999950000374997f

__device__ __forceinline__ v8f wmma_f16(v16h a, v16h b, v8f c) {
  // D = A(16x32 f16) * B(32x16 f16) + C(16x16 f32)
  return __builtin_amdgcn_wmma_f32_16x16x32_f16(
      /*neg_a=*/false, a, /*neg_b=*/false, b,
      /*c_mod=*/(short)0, c, /*reuse_a=*/false, /*reuse_b=*/false);
}

// Async DMA of a pre-built f16 LDS image from global memory into LDS,
// bypassing VGPRs (gfx1250 GLOBAL_LOAD_ASYNC_TO_LDS, ASYNCcnt-tracked).
// ldsbase = 32-bit LDS byte address, src = 16B-aligned global base,
// chunks = number of 16-byte blocks, stride = threads participating.
__device__ __forceinline__ void async_copy_to_lds(unsigned ldsbase,
                                                  const _Float16* src,
                                                  int chunks, int tid,
                                                  int stride) {
  for (int i = tid; i < chunks; i += stride) {
    unsigned off = (unsigned)i * 16u;
    asm volatile("global_load_async_to_lds_b128 %0, %1, %2"
                 :: "v"(ldsbase + off), "v"(off), "s"(src)
                 : "memory");
  }
}
__device__ __forceinline__ void wait_async_lds() {
  asm volatile("s_wait_asynccnt 0" ::: "memory");
}

// ---------------------------------------------------------------------------
// Pre-pass: build padded, transposed f16 weight images (the exact LDS image
// later DMA'd by every block). img[c][kp] (pitch 136) = W[k][c], pad = 0.
// ---------------------------------------------------------------------------
__global__ __launch_bounds__(256) void convert_wt(const float* __restrict__ W,
                                                  _Float16* __restrict__ img) {
  int i = blockIdx.x * 256 + threadIdx.x;  // over 128*136
  if (i >= WIMG_HALVES) return;
  int c = i / WPITCH, kp = i - c * WPITCH;
  img[i] = (_Float16)((kp < CDIM) ? W[kp * CDIM + c] : 0.f);
}

// Ww1 (128x8) -> transposed image [16][128], cols 8..15 zero.
__global__ __launch_bounds__(256) void convert_ww1(const float* __restrict__ W,
                                                   _Float16* __restrict__ img) {
  int i = blockIdx.x * 256 + threadIdx.x;  // over 16*128
  if (i >= 16 * CDIM) return;
  int c = i >> 7, k = i & (CDIM - 1);
  img[i] = (_Float16)((c < GRP) ? W[k * GRP + c] : 0.f);
}

// ---------------------------------------------------------------------------
// Projection GEMM: Y = act(X @ W + b). Block = 256 thr = 8 waves computes a
// 128x128 tile; wave owns a 16-row stripe, loops 8 col tiles x 4 k32 WMMAs.
// Weight image is DMA'd global->LDS asynchronously (no VGPR round trip).
// ---------------------------------------------------------------------------
template <bool BNRELU>
__global__ __launch_bounds__(256) void proj_gemm_bn(
    const float* __restrict__ X, const _Float16* __restrict__ wimg,
    const float* __restrict__ bias, const float* __restrict__ gamma,
    const float* __restrict__ beta, float* __restrict__ Y) {
  __shared__ __align__(16) _Float16 sWt[CDIM][WPITCH];

  const int tid = threadIdx.x;
  async_copy_to_lds((unsigned)(size_t)(void*)&sWt[0][0], wimg,
                    WIMG_HALVES * 2 / 16, tid, 256);

  const int wave = tid >> 5, lane = tid & 31;
  const int row = lane & 15, half = lane >> 4;
  const long rowBase = (long)blockIdx.x * 128 + wave * 16;

  // A fragments (16-bit A layout: lane = row M; two 8-wide K runs per half).
  const float4* x4 = (const float4*)(X + (rowBase + row) * CDIM);
  v16h afrag[4];
#pragma unroll
  for (int kc = 0; kc < 4; ++kc) {
    const int q0 = (kc * 32 + half * 8) >> 2;
    float4 r0 = x4[q0], r1 = x4[q0 + 1];      // K = kc*32 + 8h + 0..7
    float4 r2 = x4[q0 + 4], r3 = x4[q0 + 5];  // K = kc*32 + 16 + 8h + 0..7
    afrag[kc][0] = (_Float16)r0.x;  afrag[kc][1] = (_Float16)r0.y;
    afrag[kc][2] = (_Float16)r0.z;  afrag[kc][3] = (_Float16)r0.w;
    afrag[kc][4] = (_Float16)r1.x;  afrag[kc][5] = (_Float16)r1.y;
    afrag[kc][6] = (_Float16)r1.z;  afrag[kc][7] = (_Float16)r1.w;
    afrag[kc][8] = (_Float16)r2.x;  afrag[kc][9] = (_Float16)r2.y;
    afrag[kc][10] = (_Float16)r2.z; afrag[kc][11] = (_Float16)r2.w;
    afrag[kc][12] = (_Float16)r3.x; afrag[kc][13] = (_Float16)r3.y;
    afrag[kc][14] = (_Float16)r3.z; afrag[kc][15] = (_Float16)r3.w;
  }

  wait_async_lds();
  __syncthreads();

#pragma unroll
  for (int ct = 0; ct < 8; ++ct) {
    v8f acc = {};
#pragma unroll
    for (int kc = 0; kc < 4; ++kc) {
      v16h b;
      const _Float16* wp = &sWt[ct * 16 + row][kc * 32 + half * 8];
#pragma unroll
      for (int t = 0; t < 16; ++t) b[t] = wp[t + (t & 8)];
      acc = wmma_f16(afrag[kc], b, acc);
    }
    const int col = ct * 16 + row;
    float bb = bias[col];
    float g = 0.f, be = 0.f;
    if (BNRELU) { g = gamma[col] * BN_INV; be = beta[col]; }
    float* yp = Y + rowBase * CDIM + col;
#pragma unroll
    for (int v = 0; v < 8; ++v) {  // C/D layout: M = v + 8*half, N = row
      float val = acc[v] + bb;
      if (BNRELU) val = fmaxf(val * g + be, 0.f);
      yp[(long)(v + 8 * half) * CDIM] = val;
    }
  }
}

// ---------------------------------------------------------------------------
// Fused KNN grouped-vector attention. Block = 128 thr = 4 waves; wave = query.
// Weight images DMA'd async to LDS; neighbor k/v rows prefetched right after
// the indices are known (gathers are L2-resident, 192MB L2 holds k+v).
// ---------------------------------------------------------------------------
__global__ __launch_bounds__(128) void knn_attn_fused(
    const float* __restrict__ qf, const float* __restrict__ kf,
    const float* __restrict__ vf, const float* __restrict__ qcoord,
    const float* __restrict__ ccoord,
    const _Float16* __restrict__ wp2img, const _Float16* __restrict__ ww1img,
    const float* __restrict__ Wp1, const float* __restrict__ bp1,
    const float* __restrict__ gp1, const float* __restrict__ betap1,
    const float* __restrict__ bp2,
    const float* __restrict__ bw1, const float* __restrict__ gw1,
    const float* __restrict__ betaw1,
    const float* __restrict__ Ww2, const float* __restrict__ bw2,
    const int* __restrict__ knn, float* __restrict__ out) {
  __shared__ __align__(16) _Float16 sWp2t[CDIM][WPITCH];  // Wp2^T f16 image
  __shared__ __align__(16) _Float16 sWw1t[16][CDIM];      // Ww1^T padded 8->16
  __shared__ _Float16 sbuf[4][KNN][WPITCH];  // per-wave hidden/rel staging
  __shared__ float swb[4][KNN][GRP];         // per-wave attention weights
  __shared__ float sP1[3][CDIM];
  __shared__ float shsc[CDIM], shsh[CDIM], shb[CDIM], sbp2[CDIM];
  __shared__ float sW2[GRP][GRP];
  __shared__ float swsc[GRP], swsh[GRP], sbw1[GRP], sbw2[GRP];
  __shared__ int   sidx[4][KNN];
  __shared__ float smf[4][KNN];
  __shared__ float spos[4][KNN][3];

  const int tid = threadIdx.x;

  // Kick off the weight DMA first so it overlaps everything below.
  async_copy_to_lds((unsigned)(size_t)(void*)&sWp2t[0][0], wp2img,
                    WIMG_HALVES * 2 / 16, tid, 128);
  async_copy_to_lds((unsigned)(size_t)(void*)&sWw1t[0][0], ww1img,
                    16 * CDIM * 2 / 16, tid, 128);

  if (tid < CDIM) {
    shb[tid] = bp1[tid];
    shsc[tid] = gp1[tid] * BN_INV;
    shsh[tid] = betap1[tid];
    sbp2[tid] = bp2[tid];
    sP1[0][tid] = Wp1[tid];
    sP1[1][tid] = Wp1[CDIM + tid];
    sP1[2][tid] = Wp1[2 * CDIM + tid];
  }
  if (tid < GRP * GRP) sW2[tid >> 3][tid & 7] = Ww2[tid];
  if (tid < GRP) {
    swsc[tid] = gw1[tid] * BN_INV;
    swsh[tid] = betaw1[tid];
    sbw1[tid] = bw1[tid];
    sbw2[tid] = bw2[tid];
  }

  const int wave = tid >> 5, lane = tid & 31;
  const int row = lane & 15, half = lane >> 4;
  const long m = (long)blockIdx.x * 4 + wave;

  if (lane < KNN) {
    int ii = knn[m * KNN + lane];
    float mf = (ii >= 0) ? 1.f : 0.f;
    int ic = (ii >= 0) ? ii : 0;
    sidx[wave][lane] = ic;
    smf[wave][lane] = mf;
    spos[wave][lane][0] = (ccoord[(long)ic * 3 + 0] - qcoord[m * 3 + 0]) * mf;
    spos[wave][lane][1] = (ccoord[(long)ic * 3 + 1] - qcoord[m * 3 + 1]) * mf;
    spos[wave][lane][2] = (ccoord[(long)ic * 3 + 2] - qcoord[m * 3 + 2]) * mf;
    // Warm the caches for the upcoming k/v row gathers (512B rows).
    const float* kr = kf + (long)ic * CDIM;
    const float* vr = vf + (long)ic * CDIM;
#pragma unroll
    for (int p = 0; p < 4; ++p) {
      __builtin_prefetch(kr + p * 32, 0, 0);
      __builtin_prefetch(vr + p * 32, 0, 0);
    }
  }
  __syncthreads();

  // hidden = relu(bn(pos @ Wp1 + bp1)) -> f16 LDS, 16x128 per query
  for (int j = 0; j < KNN; ++j) {
    float p0 = spos[wave][j][0], p1 = spos[wave][j][1], p2 = spos[wave][j][2];
#pragma unroll
    for (int cb = 0; cb < 4; ++cb) {
      int c = lane + cb * 32;
      float h = p0 * sP1[0][c] + p1 * sP1[1][c] + p2 * sP1[2][c] + shb[c];
      sbuf[wave][j][c] = (_Float16)fmaxf(h * shsc[c] + shsh[c], 0.f);
    }
  }
  wait_async_lds();  // weight images resident before WMMA consumes them
  __syncthreads();

  // peb = hidden @ Wp2 + bp2 (registers, C/D layout: lane row=col, v=row M)
  v16h afrag[4];
#pragma unroll
  for (int kc = 0; kc < 4; ++kc) {
    const _Float16* hp = &sbuf[wave][row][0];
    int k0 = kc * 32 + half * 8;
#pragma unroll
    for (int t = 0; t < 8; ++t) {
      afrag[kc][t] = hp[k0 + t];
      afrag[kc][8 + t] = hp[k0 + 16 + t];
    }
  }
  v8f peb[8];
#pragma unroll
  for (int ct = 0; ct < 8; ++ct) {
    v8f acc = {};
#pragma unroll
    for (int kc = 0; kc < 4; ++kc) {
      v16h b;
      const _Float16* wp = &sWp2t[ct * 16 + row][kc * 32 + half * 8];
#pragma unroll
      for (int t = 0; t < 16; ++t) b[t] = wp[t + (t & 8)];
      acc = wmma_f16(afrag[kc], b, acc);
    }
    float bb = sbp2[ct * 16 + row];
#pragma unroll
    for (int v = 0; v < 8; ++v) peb[ct][v] = acc[v] + bb;
  }
  __syncthreads();

  // rel = k[idx] * mf - q + peb -> restage f16 (overwrites hidden)
#pragma unroll
  for (int ct = 0; ct < 8; ++ct) {
    int c = ct * 16 + row;
    float qv = qf[m * CDIM + c];
#pragma unroll
    for (int v = 0; v < 8; ++v) {
      int j = v + 8 * half;
      long ii = sidx[wave][j];
      float kg = kf[ii * CDIM + c] * smf[wave][j];
      sbuf[wave][j][c] = (_Float16)(kg - qv + peb[ct][v]);
    }
  }
  __syncthreads();

  // w1 = relu(bn(rel @ Ww1 + bw1)), one 16x16 WMMA tile (cols 8..15 zero)
  v8f wacc = {};
#pragma unroll
  for (int kc = 0; kc < 4; ++kc) {
    v16h a, b;
    const _Float16* rp = &sbuf[wave][row][0];
    int k0 = kc * 32 + half * 8;
#pragma unroll
    for (int t = 0; t < 8; ++t) {
      a[t] = rp[k0 + t];
      a[8 + t] = rp[k0 + 16 + t];
    }
    const _Float16* wp = &sWw1t[row][kc * 32 + half * 8];
#pragma unroll
    for (int t = 0; t < 16; ++t) b[t] = wp[t + (t & 8)];
    wacc = wmma_f16(a, b, wacc);
  }
  if (row < GRP) {
#pragma unroll
    for (int v = 0; v < 8; ++v)
      swb[wave][v + 8 * half][row] =
          fmaxf((wacc[v] + sbw1[row]) * swsc[row] + swsh[row], 0.f);
  }
  __syncthreads();

  // w2 = w1 @ Ww2 + bw2 ; softmax over the 16 neighbors; * mask
  float w2v[8];
  if (row < GRP) {
#pragma unroll
    for (int v = 0; v < 8; ++v) {
      int j = v + 8 * half;
      float s = sbw2[row];
#pragma unroll
      for (int g2 = 0; g2 < GRP; ++g2) s += swb[wave][j][g2] * sW2[g2][row];
      w2v[v] = s;
    }
  } else {
#pragma unroll
    for (int v = 0; v < 8; ++v) w2v[v] = -1e30f;
  }
  float mx = -1e30f;
#pragma unroll
  for (int v = 0; v < 8; ++v) mx = fmaxf(mx, w2v[v]);
  mx = fmaxf(mx, __shfl_xor(mx, 16));
  float se = 0.f;
#pragma unroll
  for (int v = 0; v < 8; ++v) { w2v[v] = __expf(w2v[v] - mx); se += w2v[v]; }
  se += __shfl_xor(se, 16);
  float rse = 1.f / se;
  __syncthreads();  // all swb reads done before overwrite
  if (row < GRP) {
#pragma unroll
    for (int v = 0; v < 8; ++v) {
      int j = v + 8 * half;
      swb[wave][j][row] = w2v[v] * rse * smf[wave][j];
    }
  }
  __syncthreads();

  // out[m][g*16 + i] = sum_j w[j][g] * (v[idx]*mf + peb); group g == tile ct
#pragma unroll
  for (int ct = 0; ct < 8; ++ct) {
    int c = ct * 16 + row;
    float s = 0.f;
#pragma unroll
    for (int v = 0; v < 8; ++v) {
      int j = v + 8 * half;
      long ii = sidx[wave][j];
      float vg = vf[ii * CDIM + c] * smf[wave][j] + peb[ct][v];
      s += swb[wave][j][ct] * vg;
    }
    s += __shfl_xor(s, 16);
    if (half == 0) out[m * CDIM + c] = s;
  }
}

// ---------------------------------------------------------------------------
extern "C" void kernel_launch(void* const* d_in, const int* in_sizes, int n_in,
                              void* d_out, int out_size, void* d_ws,
                              size_t ws_size, hipStream_t stream) {
  (void)in_sizes; (void)n_in; (void)out_size; (void)ws_size;
  const float* query_feat    = (const float*)d_in[0];
  const float* context_feat  = (const float*)d_in[1];
  const float* query_coord   = (const float*)d_in[2];
  const float* context_coord = (const float*)d_in[3];
  const float* Wq = (const float*)d_in[4];
  const float* bq = (const float*)d_in[5];
  const float* gq = (const float*)d_in[6];
  const float* betaq = (const float*)d_in[7];
  const float* Wk = (const float*)d_in[8];
  const float* bk = (const float*)d_in[9];
  const float* gk = (const float*)d_in[10];
  const float* betak = (const float*)d_in[11];
  const float* Wv = (const float*)d_in[12];
  const float* bv = (const float*)d_in[13];
  const float* Wp1 = (const float*)d_in[14];
  const float* bp1 = (const float*)d_in[15];
  const float* gp1 = (const float*)d_in[16];
  const float* betap1 = (const float*)d_in[17];
  const float* Wp2 = (const float*)d_in[18];
  const float* bp2 = (const float*)d_in[19];
  const float* Ww1 = (const float*)d_in[20];
  const float* bw1 = (const float*)d_in[21];
  const float* gw1 = (const float*)d_in[22];
  const float* betaw1 = (const float*)d_in[23];
  const float* Ww2 = (const float*)d_in[24];
  const float* bw2 = (const float*)d_in[25];
  const int*   knn = (const int*)d_in[26];

  // Workspace: q (M*C) | k (N*C) | v (N*C) fp32, then f16 weight images.
  float* q = (float*)d_ws;
  float* k = q + (size_t)M_Q * CDIM;
  float* v = k + (size_t)N_CTX * CDIM;
  _Float16* wtq  = (_Float16*)(v + (size_t)N_CTX * CDIM);
  _Float16* wtk  = wtq + WIMG_HALVES;
  _Float16* wtv  = wtk + WIMG_HALVES;
  _Float16* wtp2 = wtv + WIMG_HALVES;
  _Float16* ww1i = wtp2 + WIMG_HALVES;

  // Pre-pass: build padded transposed f16 weight images (tiny).
  convert_wt<<<WIMG_HALVES / 256, 256, 0, stream>>>(Wq, wtq);
  convert_wt<<<WIMG_HALVES / 256, 256, 0, stream>>>(Wk, wtk);
  convert_wt<<<WIMG_HALVES / 256, 256, 0, stream>>>(Wv, wtv);
  convert_wt<<<WIMG_HALVES / 256, 256, 0, stream>>>(Wp2, wtp2);
  convert_ww1<<<16 * CDIM / 256, 256, 0, stream>>>(Ww1, ww1i);

  proj_gemm_bn<true><<<M_Q / 128, 256, 0, stream>>>(query_feat, wtq, bq, gq,
                                                    betaq, q);
  proj_gemm_bn<true><<<N_CTX / 128, 256, 0, stream>>>(context_feat, wtk, bk,
                                                      gk, betak, k);
  proj_gemm_bn<false><<<N_CTX / 128, 256, 0, stream>>>(context_feat, wtv, bv,
                                                       nullptr, nullptr, v);
  knn_attn_fused<<<M_Q / 4, 128, 0, stream>>>(
      q, k, v, query_coord, context_coord, wtp2, ww1i, Wp1, bp1, gp1, betap1,
      bp2, bw1, gw1, betaw1, Ww2, bw2, knn, (float*)d_out);
}
